// MMFiModel_22651657519131
// MI455X (gfx1250) — matrix operand order repaired and checked by reading, not compiled, not run on hardware
//
#include <hip/hip_runtime.h>
#include <hip/hip_bf16.h>
#include <cstddef>

typedef __attribute__((ext_vector_type(16))) _Float16 v16h;
typedef __attribute__((ext_vector_type(8)))  _Float16 v8h;
typedef __attribute__((ext_vector_type(4)))  _Float16 v4h;
typedef __attribute__((ext_vector_type(8)))  float    v8f;

constexpr int BK = 8192;   // batch
constexpr int FK = 16;     // features
constexpr int DK = 16;     // emb dim
constexpr int VK = 100000; // vocab
constexpr int PKP = 120;   // pairs
constexpr int HK = 14;     // hidden
constexpr float BN_SCALE = 0.9999950000374997f;

// ---------------------------------------------------------------------------
// Kernel 1: gather embeddings for 16 batch rows per block, emit lin rows (f16)
// and all 120 pairwise cat products (f16).  This is the only HBM-bound stage.
// ---------------------------------------------------------------------------
__global__ void k_gather_cat(const int* __restrict__ x,
                             const float* __restrict__ embT,
                             const float* __restrict__ linT,
                             const float* __restrict__ iw,      // (P,)
                             _Float16* __restrict__ catF,       // [B][120][16]
                             _Float16* __restrict__ linF)       // [B][16][16]
{
    __shared__ float sEmb[16][16][16];
    __shared__ float sIw[PKP];
    const int b0 = blockIdx.x * 16;
    const int bl = threadIdx.x >> 4;
    const int f  = threadIdx.x & 15;

    const int idx = x[(b0 + bl) * FK + f];
    const float* er = embT + ((size_t)f * VK + idx) * DK;
    const float* lr = linT + ((size_t)f * VK + idx) * DK;

    v16h lh;
    #pragma unroll
    for (int d = 0; d < 16; ++d) {
        sEmb[bl][f][d] = er[d];
        lh[d] = (_Float16)lr[d];
    }
    *(v16h*)(linF + ((size_t)(b0 + bl) * FK + f) * DK) = lh;

    for (int i = threadIdx.x; i < PKP; i += 256) sIw[i] = iw[i];
    __syncthreads();

    for (int w = threadIdx.x; w < 16 * PKP; w += 256) {
        const int blc = w / PKP;
        const int p   = w % PKP;
        int pi = 0, rem = p;
        while (rem >= 15 - pi) { rem -= 15 - pi; ++pi; }  // triu_indices(16,1) order
        const int qi = pi + 1 + rem;
        const float s = sIw[p];
        v16h c;
        #pragma unroll
        for (int d = 0; d < 16; ++d)
            c[d] = (_Float16)(sEmb[blc][pi][d] * sEmb[blc][qi][d] * s);
        *(v16h*)(catF + ((size_t)(b0 + blc) * PKP + p) * DK) = c;
    }
}

// ---------------------------------------------------------------------------
// Kernel 2: convert attention weight matrices (s=0 slice) to f16, pad N rows.
// out layout [T][Npad][K]
// ---------------------------------------------------------------------------
__global__ void k_convW(const float* __restrict__ W, _Float16* __restrict__ out,
                        int Nvalid, int Npad, int K, int strideT)
{
    const int i = blockIdx.x * blockDim.x + threadIdx.x;
    const int total = 2 * Npad * K;
    if (i >= total) return;
    const int t = i / (Npad * K);
    const int n = (i / K) % Npad;
    const int q = i % K;
    const float v = (n < Nvalid) ? W[(size_t)t * strideT + (size_t)n * K + q] : 0.f;
    out[i] = (_Float16)v;
}

// ---------------------------------------------------------------------------
// Kernel 3: per-(b,n) branch math (s=0 only):
//   ein(14) -> share(16) / spec(8 per t) / gate logits(6 per t) -> softmax ->
//   mixed(4 per t) -> flat f16.
// All weights are tiny; staged in LDS once per block.
// ---------------------------------------------------------------------------
__global__ void k_branch_mix(const _Float16* __restrict__ feat, // [B][N][16]
                             int N,
                             const float* __restrict__ sbW,  // (14,16)
                             const float* __restrict__ sbb,  // (14,)
                             const float* __restrict__ seW,  // (4,4,14)
                             const float* __restrict__ seb,  // (16,)
                             const float* __restrict__ spW,  // s=0: (2,2,4,14)
                             const float* __restrict__ spb,  // s=0: (2,2,4)
                             const float* __restrict__ gW,   // s=0: (2,6,32)
                             const float* __restrict__ gb,   // s=0: (2,6)
                             const float* __restrict__ scen, // (16,)
                             const float* __restrict__ task, // (2,16)
                             const float* __restrict__ stw,  // (2,4,1)
                             _Float16* __restrict__ flat)    // [T][B][N*4]
{
    __shared__ float s_sbW[14 * 16];
    __shared__ float s_sbb[14];
    __shared__ float s_seW[16 * 14];
    __shared__ float s_seb[16];
    __shared__ float s_spW[16 * 14];
    __shared__ float s_spb[16];
    __shared__ float s_gW[12 * 16];   // feat half of g_W (d = 16..31)
    __shared__ float s_ct[12];        // cterm[t*6+k]

    for (int i = threadIdx.x; i < 14 * 16; i += blockDim.x) s_sbW[i] = sbW[i];
    for (int i = threadIdx.x; i < 14;      i += blockDim.x) s_sbb[i] = sbb[i];
    for (int i = threadIdx.x; i < 16 * 14; i += blockDim.x) s_seW[i] = seW[i];
    for (int i = threadIdx.x; i < 16;      i += blockDim.x) s_seb[i] = seb[i];
    for (int i = threadIdx.x; i < 16 * 14; i += blockDim.x) s_spW[i] = spW[i];
    for (int i = threadIdx.x; i < 16;      i += blockDim.x) s_spb[i] = spb[i];
    for (int i = threadIdx.x; i < 12 * 16; i += blockDim.x)
        s_gW[i] = gW[(i / 16) * 32 + 16 + (i % 16)];
    if (threadIdx.x < 12) {
        const int t = threadIdx.x / 6, k = threadIdx.x % 6;
        const float w = stw[t * 4];          // st_weights[t,0,0]
        float s = gb[t * 6 + k];
        for (int d = 0; d < 16; ++d)
            s += scen[d] * task[t * 16 + d] * w * gW[(t * 6 + k) * 32 + d];
        s_ct[threadIdx.x] = s;
    }
    __syncthreads();

    const int tid = blockIdx.x * blockDim.x + threadIdx.x;
    if (tid >= BK * N) return;
    const int b = tid / N;
    const int n = tid % N;

    float f[16];
    {
        v16h fh = *(const v16h*)(feat + (size_t)tid * 16);
        #pragma unroll
        for (int d = 0; d < 16; ++d) f[d] = (float)fh[d];
    }
    float ein[14];
    #pragma unroll
    for (int h = 0; h < 14; ++h) {
        float s = s_sbb[h];
        #pragma unroll
        for (int d = 0; d < 16; ++d) s += f[d] * s_sbW[h * 16 + d];
        ein[h] = s;
    }
    float share[16];
    #pragma unroll
    for (int j = 0; j < 16; ++j) {
        float s = s_seb[j];
        #pragma unroll
        for (int h = 0; h < 14; ++h) s += ein[h] * s_seW[j * 14 + h];
        share[j] = s;
    }

    #pragma unroll
    for (int t = 0; t < 2; ++t) {
        float spec[8];
        #pragma unroll
        for (int j = 0; j < 8; ++j) {
            float s = s_spb[t * 8 + j];
            #pragma unroll
            for (int h = 0; h < 14; ++h) s += ein[h] * s_spW[(t * 8 + j) * 14 + h];
            spec[j] = s;
        }
        float gl[6], gmx = -1e30f;
        #pragma unroll
        for (int k = 0; k < 6; ++k) {
            float s = s_ct[t * 6 + k];
            const float* gw = &s_gW[(t * 6 + k) * 16];
            #pragma unroll
            for (int d = 0; d < 16; ++d) s += f[d] * gw[d];
            gl[k] = s;
            gmx = fmaxf(gmx, s);
        }
        float gs = 0.f;
        #pragma unroll
        for (int k = 0; k < 6; ++k) { gl[k] = __expf(gl[k] - gmx); gs += gl[k]; }
        const float inv = 1.f / gs;

        v4h mh;
        #pragma unroll
        for (int m = 0; m < 4; ++m) {
            float s = 0.f;
            #pragma unroll
            for (int e = 0; e < 4; ++e)  s += gl[e]     * share[e * 4 + m];
            #pragma unroll
            for (int e2 = 0; e2 < 2; ++e2) s += gl[4 + e2] * spec[e2 * 4 + m];
            mh[m] = (_Float16)(s * inv);
        }
        *(v4h*)(flat + ((size_t)t * BK + b) * (size_t)(N * 4) + n * 4) = mh;
    }
}

// ---------------------------------------------------------------------------
// Kernel 4: attention-logit GEMM with v_wmma_f32_16x16x32_f16.
//   logits[t][b][n] = flat[t][b][:K] . W[t][n][:K] + bias[t][n]
// One wave handles a 16-row b-tile x one 16-col n-tile; K looped by 32.
// A fragment (16x32 f16): lanes 0-15 hold K 0-15, lanes 16-31 hold K 16-31
// (same k order as B) -> two contiguous 16B loads per lane.
// B fragment (32x16): lane = n + 16*(k/16), halfs = k%16 -> one contiguous
// 32B load per lane from weight row n (B(k,n) = W[n][k], no transpose).
// ---------------------------------------------------------------------------
__global__ void k_wmma_logits(const _Float16* __restrict__ flat, // [T][B][K]
                              const _Float16* __restrict__ W,    // [T][Npad][K]
                              const float* __restrict__ bias,    // + t*biasStrideT
                              float* __restrict__ out,           // [T][B][Nvalid]
                              int K, int Npad, int Nvalid, int biasStrideT)
{
    const int lane = threadIdx.x & 31;
    const int wave = threadIdx.x >> 5;
    const int b0 = (blockIdx.x * (blockDim.x >> 5) + wave) * 16;
    const int n0 = blockIdx.y * 16;
    const int t  = blockIdx.z;

    const int l15 = lane & 15;
    const int hi  = lane & 16;                 // 0 or 16

    const _Float16* aBase = flat + ((size_t)t * BK + (b0 + l15)) * (size_t)K + hi;
    const _Float16* bBase = W    + ((size_t)t * Npad + (n0 + l15)) * (size_t)K + hi;

    v8f acc = {};
    for (int k0 = 0; k0 < K; k0 += 32) {
        v8h alo = *(const v8h*)(aBase + k0);        // K rel (base..base+7)
        v8h ahi = *(const v8h*)(aBase + k0 + 8);    // K rel (base+8..base+15)
        v16h a;
        #pragma unroll
        for (int i = 0; i < 8; ++i) { a[i] = alo[i]; a[8 + i] = ahi[i]; }
        v16h b = *(const v16h*)(bBase + k0);        // 16 consecutive K for col n
        acc = __builtin_amdgcn_wmma_f32_16x16x32_f16(
            false, a, false, b, (short)0, acc, false, false);
    }

    const int n = n0 + l15;
    if (n < Nvalid) {
        const float bv = bias[t * biasStrideT + n];
        const int mBase = hi >> 1;                  // 0 or 8
        #pragma unroll
        for (int r = 0; r < 8; ++r)
            out[((size_t)t * BK + (b0 + mBase + r)) * (size_t)Nvalid + n] = acc[r] + bv;
    }
}

// ---------------------------------------------------------------------------
// Kernel 5: softmax over attention logits + attended-vector reduction.
// Only sum_d of the attended vector is needed, so the contraction collapses
// to sum_n softmax(l)_n * rowsum(feat_n).  + bias_p, sigmoid, write (B,2).
// ---------------------------------------------------------------------------
__global__ void k_final(const float* __restrict__ logF,    // [T][B][120]
                        const float* __restrict__ logL,    // [T][B][16]
                        const _Float16* __restrict__ catF, // [B][120][16]
                        const _Float16* __restrict__ linF, // [B][16][16]
                        const float* __restrict__ bias_p,  // (T,4,16)
                        float* __restrict__ out)           // [B][T]
{
    const int tid = blockIdx.x * blockDim.x + threadIdx.x;
    if (tid >= 2 * BK) return;
    const int t = tid >> 13;
    const int b = tid & (BK - 1);

    float logit = 0.f;

    { // fm branch, N = 120
        const float* lg = logF + ((size_t)t * BK + b) * PKP;
        float mx = -1e30f;
        for (int n = 0; n < PKP; ++n) mx = fmaxf(mx, lg[n] * BN_SCALE);
        float se = 0.f, acc = 0.f;
        for (int n = 0; n < PKP; ++n) {
            const float w = __expf(lg[n] * BN_SCALE - mx);
            v16h r = *(const v16h*)(catF + ((size_t)b * PKP + n) * DK);
            float rs = 0.f;
            #pragma unroll
            for (int d = 0; d < 16; ++d) rs += (float)r[d];
            se += w; acc += w * rs;
        }
        logit += acc / se;
    }
    { // lin branch, N = 16
        const float* lg = logL + ((size_t)t * BK + b) * FK;
        float mx = -1e30f;
        for (int n = 0; n < FK; ++n) mx = fmaxf(mx, lg[n] * BN_SCALE);
        float se = 0.f, acc = 0.f;
        for (int n = 0; n < FK; ++n) {
            const float w = __expf(lg[n] * BN_SCALE - mx);
            v16h r = *(const v16h*)(linF + ((size_t)b * FK + n) * DK);
            float rs = 0.f;
            #pragma unroll
            for (int d = 0; d < 16; ++d) rs += (float)r[d];
            se += w; acc += w * rs;
        }
        logit += acc / se;
    }
    float bsum = 0.f;
    #pragma unroll
    for (int d = 0; d < 16; ++d) bsum += bias_p[t * 64 + d];   // bias_p[t][s=0][d]
    logit += bsum;

    out[b * 2 + t] = 1.f / (1.f + __expf(-logit));
}

// ---------------------------------------------------------------------------
extern "C" void kernel_launch(void* const* d_in, const int* in_sizes, int n_in,
                              void* d_out, int out_size, void* d_ws, size_t ws_size,
                              hipStream_t stream) {
    (void)in_sizes; (void)n_in; (void)out_size; (void)ws_size;

    const int*   x      = (const int*)  d_in[0];
    const float* embT   = (const float*)d_in[1];
    const float* linT   = (const float*)d_in[2];
    const float* scen   = (const float*)d_in[3];
    const float* task   = (const float*)d_in[4];
    const float* stw    = (const float*)d_in[5];
    const float* iw     = (const float*)d_in[6];
    const float* sb_W   = (const float*)d_in[7];
    const float* sb_b   = (const float*)d_in[8];
    const float* lsb_W  = (const float*)d_in[9];
    const float* lsb_b  = (const float*)d_in[10];
    const float* se_W   = (const float*)d_in[11];
    const float* se_b   = (const float*)d_in[12];
    const float* lse_W  = (const float*)d_in[13];
    const float* lse_b  = (const float*)d_in[14];
    const float* sp_W   = (const float*)d_in[15];  // s=0 slice == base
    const float* sp_b   = (const float*)d_in[16];
    const float* lsp_W  = (const float*)d_in[17];
    const float* lsp_b  = (const float*)d_in[18];
    const float* g_W    = (const float*)d_in[19];
    const float* g_b    = (const float*)d_in[20];
    const float* lg_W   = (const float*)d_in[21];
    const float* lg_b   = (const float*)d_in[22];
    const float* at_W   = (const float*)d_in[23];
    const float* at_b   = (const float*)d_in[24];
    const float* lt_W   = (const float*)d_in[25];
    const float* lt_b   = (const float*)d_in[26];
    const float* bias_p = (const float*)d_in[27];
    float* out = (float*)d_out;

    // ---- workspace carving (all 256B aligned; ~63 MB total) ----
    char* p = (char*)d_ws;
    auto carve = [&](size_t bytes) {
        void* r = (void*)p;
        p += (bytes + 255) & ~(size_t)255;
        return r;
    };
    _Float16* catF  = (_Float16*)carve((size_t)BK * PKP * DK * 2);   // 31.5 MB
    _Float16* linF  = (_Float16*)carve((size_t)BK * FK  * DK * 2);   //  4.2 MB
    _Float16* flatF = (_Float16*)carve((size_t)2 * BK * 480 * 2);    // 15.7 MB
    _Float16* flatL = (_Float16*)carve((size_t)2 * BK * 64  * 2);    //  2.1 MB
    _Float16* WF    = (_Float16*)carve((size_t)2 * 128 * 480 * 2);   //  0.25 MB
    _Float16* WL    = (_Float16*)carve((size_t)2 * 16  * 64  * 2);   //  4 KB
    float*    logF  = (float*)   carve((size_t)2 * BK * PKP * 4);    //  7.9 MB
    float*    logL  = (float*)   carve((size_t)2 * BK * FK  * 4);    //  1.0 MB

    // 1) gather + pairwise products
    k_gather_cat<<<BK / 16, 256, 0, stream>>>(x, embT, linT, iw, catF, linF);

    // 2) weight conversion (s=0 slices)
    k_convW<<<(2 * 128 * 480 + 255) / 256, 256, 0, stream>>>(
        at_W, WF, 120, 128, 480, 120 * 480);
    k_convW<<<(2 * 16 * 64 + 255) / 256, 256, 0, stream>>>(
        lt_W, WL, 16, 16, 64, 16 * 64);

    // 3) branch mix (fm: N=120, lin: N=16)
    k_branch_mix<<<(BK * PKP) / 256, 256, 0, stream>>>(
        catF, PKP, sb_W, sb_b, se_W, se_b, sp_W, sp_b, g_W, g_b,
        scen, task, stw, flatF);
    k_branch_mix<<<(BK * FK) / 256, 256, 0, stream>>>(
        linF, FK, lsb_W, lsb_b, lse_W, lse_b, lsp_W, lsp_b, lg_W, lg_b,
        scen, task, stw, flatL);

    // 4) WMMA attention-logit GEMMs
    //    fm : M=8192, K=480, N=120 (pad 128), per t
    //    lin: M=8192, K=64,  N=16,            per t
    k_wmma_logits<<<dim3(BK / 64, 128 / 16, 2), 128, 0, stream>>>(
        flatF, WF, at_b, logF, 480, 128, 120, 120);
    k_wmma_logits<<<dim3(BK / 64, 1, 2), 128, 0, stream>>>(
        flatL, WL, lt_b, logL, 64, 16, 16, 16);

    // 5) softmax + attended reduction + sigmoid
    k_final<<<(2 * BK) / 256, 256, 0, stream>>>(
        logF, logL, catF, linF, bias_p, out);
}
